// RNN_28707561406830
// MI455X (gfx1250) — matrix operand order repaired and checked by reading, not compiled, not run on hardware
//
#include <hip/hip_runtime.h>
#include <hip/hip_bf16.h>
#include <math.h>

#define SEQ_LEN 4096
#define DATA_DIM 1024
#define HID 2048
#define GATES (4 * HID) /* 8192 */
#define LABELS 1000

typedef __bf16 bf16_t;
typedef bf16_t v8bf  __attribute__((ext_vector_type(8)));
typedef bf16_t v16bf __attribute__((ext_vector_type(16)));
typedef float  v8f   __attribute__((ext_vector_type(8)));

// ---------------------------------------------------------------------------
// K0: convert x and W_ih to bf16, zero h-buffer 0, reset grid-barrier state.
// ---------------------------------------------------------------------------
__global__ void k_prep(const float* __restrict__ x, const float* __restrict__ Wih,
                       bf16_t* __restrict__ xb, bf16_t* __restrict__ wb,
                       float* __restrict__ hbuf, unsigned* __restrict__ sync) {
  const long long tid = (long long)blockIdx.x * blockDim.x + threadIdx.x;
  const long long stride = (long long)gridDim.x * blockDim.x;
  const long long nx = (long long)SEQ_LEN * DATA_DIM;
  const long long nw = (long long)GATES * DATA_DIM;
  for (long long i = tid; i < nx; i += stride) xb[i] = (bf16_t)x[i];
  for (long long i = tid; i < nw; i += stride) wb[i] = (bf16_t)Wih[i];
  for (long long i = tid; i < HID; i += stride) hbuf[i] = 0.0f;  // h0 = 0
  if (tid == 0) { sync[0] = 0u; sync[1] = 0u; }
}

// ---------------------------------------------------------------------------
// K1: XG[t][g] = x @ W_ih^T + (b_ih + b_hh), bf16 WMMA, fp32 accumulate.
// One 16x16 output tile per wave32; K-loop of 32-wide WMMA steps.
// ---------------------------------------------------------------------------
__global__ void __launch_bounds__(256)
k_gemm_xgates(const bf16_t* __restrict__ xb, const bf16_t* __restrict__ wb,
              const float* __restrict__ b_ih, const float* __restrict__ b_hh,
              float* __restrict__ xg) {
  const int lane = threadIdx.x & 31;
  const int wave = threadIdx.x >> 5;
  const int tile = blockIdx.x * 8 + wave;
  const int NT = GATES / 16;                 // 512 tiles along gate dim
  const int m0 = (tile / NT) * 16;           // seq tile base
  const int n0 = (tile % NT) * 16;           // gate tile base
  const int hi = lane >> 4;                  // half-wave select (0/1)
  const int ln = lane & 15;

  // A: row M = m0+ln of x; lane<16 holds K {0..15}, lane>=16 {16..31}
  const bf16_t* pa = xb + (long long)(m0 + ln) * DATA_DIM + hi * 16;
  // B[k][n] = W_ih[n][k]; lane holds 16 consecutive K of row n0+ln
  const bf16_t* pb = wb + (long long)(n0 + ln) * DATA_DIM + hi * 16;

  v8f acc = {};
#pragma unroll 4
  for (int kk = 0; kk < DATA_DIM; kk += 32) {
    v8bf a_lo = *(const v8bf*)(pa + kk);        // 16B load
    v8bf a_hi = *(const v8bf*)(pa + kk + 8);    // 16B load
    v16bf a = __builtin_shufflevector(a_lo, a_hi,
                                      0, 1, 2, 3, 4, 5, 6, 7,
                                      8, 9, 10, 11, 12, 13, 14, 15);
    v16bf b = *(const v16bf*)(pb + kk);         // 32B load
    acc = __builtin_amdgcn_wmma_f32_16x16x32_bf16(
        false, a, false, b, (short)0, acc, false, false);
  }

  // C/D layout: VGPR r -> M = m0 + r + 8*hi, N = n0 + ln
  const int n = n0 + ln;
  const float bias = b_ih[n] + b_hh[n];
  const int mbase = m0 + hi * 8;
#pragma unroll
  for (int r = 0; r < 8; ++r)
    xg[(long long)(mbase + r) * GATES + n] = acc[r] + bias;
}

// ---------------------------------------------------------------------------
// K2: persistent LSTM scan. 128 blocks x 512 threads; wave w of block b owns
// h index j = b*16 + w and its 4 gate rows of W_hh. c stays in a register.
// Device-wide sense-reversing barrier via agent-scope atomics in d_ws.
// ---------------------------------------------------------------------------
#define SCAN_BLOCKS 128
#define SCAN_THREADS 512

__device__ __forceinline__ float wave_sum32(float v) {
#pragma unroll
  for (int m = 16; m > 0; m >>= 1) v += __shfl_xor(v, m, 32);
  return v;
}

__global__ void __launch_bounds__(SCAN_THREADS)
k_scan(const float* __restrict__ Whh, const float* __restrict__ xg,
       float* __restrict__ hbuf, unsigned* __restrict__ sync) {
  __shared__ float h_s[HID];
  const int lane = threadIdx.x & 31;
  const int wave = threadIdx.x >> 5;          // 0..15
  const int j = blockIdx.x * 16 + wave;       // owned h element
  const float* w0 = Whh + (long long)j * HID;               // i-gate row
  const float* w1 = w0 + (long long)HID * HID;               // f-gate row
  const float* w2 = w1 + (long long)HID * HID;               // g-gate row
  const float* w3 = w2 + (long long)HID * HID;               // o-gate row
  unsigned* cnt = sync;
  unsigned* gen = sync + 1;
  float c = 0.0f;  // cell state, live in lane 0's register

  for (int t = 0; t < SEQ_LEN; ++t) {
    const float* hin = hbuf + (t & 1) * HID;
    float* hout = hbuf + ((t + 1) & 1) * HID;

    // Stage h into LDS with device-scope loads (bypass stale WGP$).
    for (int i = threadIdx.x; i < HID; i += SCAN_THREADS)
      h_s[i] = __hip_atomic_load(&hin[i], __ATOMIC_RELAXED,
                                 __HIP_MEMORY_SCOPE_AGENT);
    __syncthreads();

    float a0 = 0.f, a1 = 0.f, a2 = 0.f, a3 = 0.f;
#pragma unroll
    for (int i = 0; i < HID / 128; ++i) {      // 16 iterations, b128 loads
      const int k = i * 128 + lane * 4;
      const float4 hv = *(const float4*)(h_s + k);
      const float4 x0 = *(const float4*)(w0 + k);
      const float4 x1 = *(const float4*)(w1 + k);
      const float4 x2 = *(const float4*)(w2 + k);
      const float4 x3 = *(const float4*)(w3 + k);
      a0 += hv.x * x0.x + hv.y * x0.y + hv.z * x0.z + hv.w * x0.w;
      a1 += hv.x * x1.x + hv.y * x1.y + hv.z * x1.z + hv.w * x1.w;
      a2 += hv.x * x2.x + hv.y * x2.y + hv.z * x2.z + hv.w * x2.w;
      a3 += hv.x * x3.x + hv.y * x3.y + hv.z * x3.z + hv.w * x3.w;
    }
    a0 = wave_sum32(a0);
    a1 = wave_sum32(a1);
    a2 = wave_sum32(a2);
    a3 = wave_sum32(a3);

    if (lane == 0) {
      const float* g = xg + (long long)t * GATES;
      float gi = a0 + g[j];
      float gf = a1 + g[j + HID];
      float gg = a2 + g[j + 2 * HID];
      float go = a3 + g[j + 3 * HID];
      gi = 1.0f / (1.0f + __expf(-gi));
      gf = 1.0f / (1.0f + __expf(-gf));
      gg = tanhf(gg);
      go = 1.0f / (1.0f + __expf(-go));
      c = gf * c + gi * gg;
      const float h = go * tanhf(c);
      __hip_atomic_store(&hout[j], h, __ATOMIC_RELAXED,
                         __HIP_MEMORY_SCOPE_AGENT);
    }
    __syncthreads();  // h_s fully consumed; block's h write issued

    // ---- device-wide barrier (sense = t+1) ----
    if (threadIdx.x == 0) {
      const unsigned prev = __hip_atomic_fetch_add(
          cnt, 1u, __ATOMIC_ACQ_REL, __HIP_MEMORY_SCOPE_AGENT);
      if (prev == SCAN_BLOCKS - 1) {
        __hip_atomic_store(cnt, 0u, __ATOMIC_RELAXED,
                           __HIP_MEMORY_SCOPE_AGENT);
        __hip_atomic_fetch_add(gen, 1u, __ATOMIC_RELEASE,
                               __HIP_MEMORY_SCOPE_AGENT);
      } else {
        while (__hip_atomic_load(gen, __ATOMIC_ACQUIRE,
                                 __HIP_MEMORY_SCOPE_AGENT) < (unsigned)(t + 1))
          __builtin_amdgcn_s_sleep(2);
      }
    }
    __syncthreads();
  }
}

// ---------------------------------------------------------------------------
// K3: out = softmax(W_out @ h_final). Single block.
// ---------------------------------------------------------------------------
__global__ void __launch_bounds__(256)
k_head(const float* __restrict__ Wout, const float* __restrict__ hfin,
       float* __restrict__ out) {
  __shared__ float h_s[HID];
  __shared__ float logits[LABELS];
  __shared__ float red[2];
  for (int i = threadIdx.x; i < HID; i += 256) h_s[i] = hfin[i];
  __syncthreads();

  const int lane = threadIdx.x & 31;
  const int wave = threadIdx.x >> 5;  // 8 waves
  for (int r = wave; r < LABELS; r += 8) {
    const float* w = Wout + (long long)r * HID;
    float a = 0.f;
#pragma unroll
    for (int i = 0; i < HID / 128; ++i) {
      const int k = i * 128 + lane * 4;
      const float4 wv = *(const float4*)(w + k);
      a += h_s[k] * wv.x + h_s[k + 1] * wv.y +
           h_s[k + 2] * wv.z + h_s[k + 3] * wv.w;
    }
    a = wave_sum32(a);
    if (lane == 0) logits[r] = a;
  }
  __syncthreads();

  if (threadIdx.x == 0) {
    float mx = -INFINITY;
    for (int i = 0; i < LABELS; ++i) mx = fmaxf(mx, logits[i]);
    float s = 0.f;
    for (int i = 0; i < LABELS; ++i) s += __expf(logits[i] - mx);
    red[0] = mx;
    red[1] = s;
  }
  __syncthreads();
  const float mx = red[0];
  const float inv = 1.0f / red[1];
  for (int i = threadIdx.x; i < LABELS; i += 256)
    out[i] = __expf(logits[i] - mx) * inv;
}

// ---------------------------------------------------------------------------
// Launch
// ---------------------------------------------------------------------------
extern "C" void kernel_launch(void* const* d_in, const int* in_sizes, int n_in,
                              void* d_out, int out_size, void* d_ws, size_t ws_size,
                              hipStream_t stream) {
  (void)in_sizes; (void)n_in; (void)out_size; (void)ws_size;
  const float* x    = (const float*)d_in[0];
  const float* Wih  = (const float*)d_in[1];
  const float* Whh  = (const float*)d_in[2];
  const float* b_ih = (const float*)d_in[3];
  const float* b_hh = (const float*)d_in[4];
  const float* Wout = (const float*)d_in[5];
  float* out = (float*)d_out;

  // Workspace layout (256B aligned chunks)
  char* ws = (char*)d_ws;
  const size_t XG_BYTES = (size_t)SEQ_LEN * GATES * sizeof(float);     // 134.2 MB
  const size_t XB_BYTES = (size_t)SEQ_LEN * DATA_DIM * sizeof(bf16_t); //   8.4 MB
  const size_t WB_BYTES = (size_t)GATES * DATA_DIM * sizeof(bf16_t);   //  16.8 MB
  float*   xg   = (float*)ws;
  bf16_t*  xb   = (bf16_t*)(ws + XG_BYTES);
  bf16_t*  wb   = (bf16_t*)(ws + XG_BYTES + XB_BYTES);
  float*   hbuf = (float*)(ws + XG_BYTES + XB_BYTES + WB_BYTES);       // 2*HID f32
  unsigned* sync = (unsigned*)(ws + XG_BYTES + XB_BYTES + WB_BYTES +
                               2 * HID * sizeof(float));

  // K0: bf16 conversion + state init
  k_prep<<<1024, 256, 0, stream>>>(x, Wih, xb, wb, hbuf, sync);

  // K1: input-projection GEMM, 131072 tiles, 8 waves/block
  const int tiles = (SEQ_LEN / 16) * (GATES / 16);
  k_gemm_xgates<<<tiles / 8, 256, 0, stream>>>(xb, wb, b_ih, b_hh, xg);

  // K2: persistent recurrent scan
  k_scan<<<SCAN_BLOCKS, SCAN_THREADS, 0, stream>>>(Whh, xg, hbuf, sync);

  // K3: softmax head; SEQ_LEN even -> final h lives in buffer 0
  k_head<<<1, 256, 0, stream>>>(Wout, hbuf, out);
}
